// EvolveGCNBlock_69526930588436
// MI455X (gfx1250) — matrix operand order repaired and checked by reading, not compiled, not run on hardware
//
#include <hip/hip_runtime.h>
#include <math.h>

// ---------------- problem constants (match reference) ----------------
constexpr int S = 4;
constexpr int N = 50000;     // multiple of 16 -> no partial WMMA tiles
constexpr int C = 256;
constexpr int E = 800000;
constexpr int L = 3;

// ---------------- WMMA types / helpers ----------------
typedef __attribute__((ext_vector_type(16))) __bf16 v16bf;
typedef __attribute__((ext_vector_type(8)))  __bf16 v8bf;
typedef __attribute__((ext_vector_type(8)))  float  v8f;

__device__ __forceinline__ unsigned short f2bf(float f) {
  unsigned int u = __float_as_uint(f);
  u += 0x7FFFu + ((u >> 16) & 1u);   // round-to-nearest-even
  return (unsigned short)(u >> 16);
}

// A-fragment (16x32 bf16, MxK): lane holds row (lane&15); k elements are one
// 16-wide run at k0 + 16*(lane>>4).  Two 16B loads.
__device__ __forceinline__ v16bf load_a_frag(const unsigned short* p) {
  v8bf lo = *(const v8bf*)(p);
  v8bf hi = *(const v8bf*)(p + 8);
  return __builtin_shufflevector(lo, hi, 0,1,2,3,4,5,6,7,8,9,10,11,12,13,14,15);
}

#define WMMA_BF16(a, b, c) \
  __builtin_amdgcn_wmma_f32_16x16x32_bf16(false, (a), false, (b), (short)0, (c), false, false)

// ---------------- small utility kernels ----------------
__global__ void fill_f32_kernel(float* __restrict__ p, long n, float v) {
  long i = (long)blockIdx.x * blockDim.x + threadIdx.x;
  if (i < n) p[i] = v;
}
__global__ void fill_u16_kernel(unsigned short* __restrict__ p, long n, unsigned short v) {
  long i = (long)blockIdx.x * blockDim.x + threadIdx.x;
  if (i < n) p[i] = v;
}
__global__ void cast_bf_kernel(const float* __restrict__ s, unsigned short* __restrict__ d, long n) {
  long i = (long)blockIdx.x * blockDim.x + threadIdx.x;
  if (i < n) d[i] = f2bf(s[i]);
}
// Wt[l][n][k] = W[l][k][n]  (B-operand layout: per column n, k contiguous)
__global__ void stage_wgcn_kernel(const float* __restrict__ W, unsigned short* __restrict__ Wt) {
  int i = blockIdx.x * blockDim.x + threadIdx.x;
  if (i >= L * C * C) return;
  int l = i / (C * C);
  int r = i % (C * C);
  int n = r / C;
  int k = r % C;
  Wt[i] = f2bf(W[(size_t)l * C * C + (size_t)k * C + n]);
}

// ---------------- graph kernels ----------------
__global__ void degree_kernel(const int* __restrict__ col, const float* __restrict__ ew,
                              float* __restrict__ deg) {
  int e = blockIdx.x * blockDim.x + threadIdx.x;
  if (e < E) atomicAdd(&deg[col[e]], ew[e]);
}
__global__ void dinv_kernel(const float* __restrict__ deg, float* __restrict__ dinv) {
  int n = blockIdx.x * blockDim.x + threadIdx.x;
  if (n < N) dinv[n] = rsqrtf(deg[n] + 1.0f);   // +1 = self loop weight
}
// 4 edges per block; 64 threads/edge, float4 (b128) gather + 4 L2 float atomics
__global__ __launch_bounds__(256)
void agg_kernel(const int* __restrict__ rowi, const int* __restrict__ coli,
                const float* __restrict__ ew, const float* __restrict__ dinv,
                const float* __restrict__ hW, float* __restrict__ agg) {
  const int e  = blockIdx.x * 4 + (threadIdx.x >> 6);
  const int c4 = (threadIdx.x & 63) << 2;
  const int r  = rowi[e];
  const int cl = coli[e];
  const float norm = dinv[r] * ew[e] * dinv[cl];
  const float4 v = *(const float4*)(hW + (size_t)r * C + c4);
  float* dst = agg + (size_t)cl * C + c4;
  atomicAdd(dst + 0, v.x * norm);
  atomicAdd(dst + 1, v.y * norm);
  atomicAdd(dst + 2, v.z * norm);
  atomicAdd(dst + 3, v.w * norm);
}

// ---------------- GCN GEMM, 64-row macro-tile ----------------
// out[64,C] = Abf[64,C] @ W (Wt layout). 8 waves, each: 4 row-tiles x 2 col-tiles
// -> 8 WMMA per 12 b128 loads per K-step (B-frags amortized over 4 row-tiles).
__global__ __launch_bounds__(256)
void gcn_gemm64_kernel(const unsigned short* __restrict__ A,
                       const unsigned short* __restrict__ Bt,
                       float* __restrict__ Out, int row_base) {
  const int wave = threadIdx.x >> 5;        // 8 waves
  const int lane = threadIdx.x & 31;
  const int hl   = lane >> 4;
  const int ln   = lane & 15;
  const int row0 = row_base + (blockIdx.x << 6);   // 64 rows per block
  const int n0   = wave * 32 + ln;                 // 2 column tiles per wave
  v8f acc00 = {}, acc01 = {}, acc10 = {}, acc11 = {};
  v8f acc20 = {}, acc21 = {}, acc30 = {}, acc31 = {};
#pragma unroll
  for (int k0 = 0; k0 < C; k0 += 32) {
    const int kb = k0 + hl * 16;
    v16bf b0 = *(const v16bf*)(Bt + (size_t)n0 * C + kb);
    v16bf b1 = *(const v16bf*)(Bt + (size_t)(n0 + 16) * C + kb);
    const size_t abase = (size_t)(row0 + ln) * C + k0 + hl * 16;
    v16bf a0 = load_a_frag(A + abase);
    v16bf a1 = load_a_frag(A + abase + (size_t)16 * C);
    v16bf a2 = load_a_frag(A + abase + (size_t)32 * C);
    v16bf a3 = load_a_frag(A + abase + (size_t)48 * C);
    acc00 = WMMA_BF16(a0, b0, acc00);  acc01 = WMMA_BF16(a0, b1, acc01);
    acc10 = WMMA_BF16(a1, b0, acc10);  acc11 = WMMA_BF16(a1, b1, acc11);
    acc20 = WMMA_BF16(a2, b0, acc20);  acc21 = WMMA_BF16(a2, b1, acc21);
    acc30 = WMMA_BF16(a3, b0, acc30);  acc31 = WMMA_BF16(a3, b1, acc31);
  }
  v8f a0[2] = {acc00, acc01}, a1[2] = {acc10, acc11};
  v8f a2[2] = {acc20, acc21}, a3[2] = {acc30, acc31};
#pragma unroll
  for (int r = 0; r < 8; ++r) {
    const int mm = row0 + r + 8 * hl;             // D layout: VGPR r -> row r+8*half
    Out[(size_t)(mm     ) * C + n0]      = a0[0][r];
    Out[(size_t)(mm     ) * C + n0 + 16] = a0[1][r];
    Out[(size_t)(mm + 16) * C + n0]      = a1[0][r];
    Out[(size_t)(mm + 16) * C + n0 + 16] = a1[1][r];
    Out[(size_t)(mm + 32) * C + n0]      = a2[0][r];
    Out[(size_t)(mm + 32) * C + n0 + 16] = a2[1][r];
    Out[(size_t)(mm + 48) * C + n0]      = a3[0][r];
    Out[(size_t)(mm + 48) * C + n0 + 16] = a3[1][r];
  }
}

// 16-row tail kernel (N/16 = 3125 tiles; 781*4 = 3124 covered by the 64-row kernel)
__global__ __launch_bounds__(256)
void gcn_gemm16_kernel(const unsigned short* __restrict__ A,
                       const unsigned short* __restrict__ Bt,
                       float* __restrict__ Out, int row_base) {
  const int wave = threadIdx.x >> 5;
  const int lane = threadIdx.x & 31;
  const int hl   = lane >> 4;
  const int ln   = lane & 15;
  const int row0 = row_base + (blockIdx.x << 4);
  const int m    = row0 + ln;
  const int n0   = wave * 32 + ln;
  v8f acc0 = {}, acc1 = {};
#pragma unroll
  for (int k0 = 0; k0 < C; k0 += 32) {
    v16bf a = load_a_frag(A + (size_t)m * C + k0 + hl * 16);
    const int kb = k0 + hl * 16;
    v16bf b0 = *(const v16bf*)(Bt + (size_t)n0 * C + kb);
    v16bf b1 = *(const v16bf*)(Bt + (size_t)(n0 + 16) * C + kb);
    acc0 = WMMA_BF16(a, b0, acc0);
    acc1 = WMMA_BF16(a, b1, acc1);
  }
#pragma unroll
  for (int r = 0; r < 8; ++r) {
    const int mm = row0 + r + 8 * hl;
    Out[(size_t)mm * C + n0]      = acc0[r];
    Out[(size_t)mm * C + n0 + 16] = acc1[r];
  }
}

// ---------------- fused self-loop + bias + LayerNorm + ReLU + bf16 cast ----------------
__global__ __launch_bounds__(256)
void ln_relu_kernel(const float* __restrict__ agg, const float* __restrict__ hW,
                    const float* __restrict__ dinv, const float* __restrict__ bias,
                    const float* __restrict__ gamma, const float* __restrict__ beta,
                    unsigned short* __restrict__ outbf) {
  const int n = blockIdx.x;
  const int c = threadIdx.x;
  const float di = dinv[n];
  const size_t idx = (size_t)n * C + c;
  const float pre = agg[idx] + hW[idx] * (di * di) + bias[c];
  __shared__ float red[C];
  red[c] = pre;
  __syncthreads();
  for (int s = 128; s > 0; s >>= 1) { if (c < s) red[c] += red[c + s]; __syncthreads(); }
  const float mu = red[0] * (1.0f / C);
  __syncthreads();
  const float d = pre - mu;
  red[c] = d * d;
  __syncthreads();
  for (int s = 128; s > 0; s >>= 1) { if (c < s) red[c] += red[c + s]; __syncthreads(); }
  const float var = red[0] * (1.0f / C);
  float y = d * rsqrtf(var + 1e-5f) * gamma[c] + beta[c];
  y = fmaxf(y, 0.0f);
  outbf[idx] = f2bf(y);
}

// ---------------- fused GRU step: both GEMMs + gate math, in place ----------------
__global__ __launch_bounds__(256)
void gru_step_kernel(const unsigned short* __restrict__ Xbf,  // feats[t], bf16
                     unsigned short* Hbf,                     // h state bf16 (read A, write new)
                     const float* Hf,                         // h state f32 (read; may alias Hout)
                     const unsigned short* __restrict__ Wih,  // [3C][C] bf16 (B-layout)
                     const unsigned short* __restrict__ Whh,  // [3C][C] bf16
                     const float* __restrict__ bih, const float* __restrict__ bhh,
                     float* Hout) {                           // f32 new h (hstate or d_out)
  const int wave = threadIdx.x >> 5;
  const int lane = threadIdx.x & 31;
  const int hl   = lane >> 4;
  const int ln   = lane & 15;
  const int row0 = blockIdx.x << 4;
  const int m    = row0 + ln;
  const int n0   = wave * 32 + ln;   // channel tile 0 for this wave
  const int n1   = n0 + 16;          // channel tile 1
  v8f air0={}, aiz0={}, ain0={}, ahr0={}, ahz0={}, ahn0={};
  v8f air1={}, aiz1={}, ain1={}, ahr1={}, ahz1={}, ahn1={};
#pragma unroll
  for (int k0 = 0; k0 < C; k0 += 32) {
    v16bf ax = load_a_frag(Xbf + (size_t)m * C + k0 + hl * 16);
    v16bf ah = load_a_frag(Hbf + (size_t)m * C + k0 + hl * 16);
    const int kb = k0 + hl * 16;
    air0 = WMMA_BF16(ax, *(const v16bf*)(Wih + (size_t)(n0        ) * C + kb), air0);
    aiz0 = WMMA_BF16(ax, *(const v16bf*)(Wih + (size_t)(n0 +     C) * C + kb), aiz0);
    ain0 = WMMA_BF16(ax, *(const v16bf*)(Wih + (size_t)(n0 + 2 * C) * C + kb), ain0);
    ahr0 = WMMA_BF16(ah, *(const v16bf*)(Whh + (size_t)(n0        ) * C + kb), ahr0);
    ahz0 = WMMA_BF16(ah, *(const v16bf*)(Whh + (size_t)(n0 +     C) * C + kb), ahz0);
    ahn0 = WMMA_BF16(ah, *(const v16bf*)(Whh + (size_t)(n0 + 2 * C) * C + kb), ahn0);
    air1 = WMMA_BF16(ax, *(const v16bf*)(Wih + (size_t)(n1        ) * C + kb), air1);
    aiz1 = WMMA_BF16(ax, *(const v16bf*)(Wih + (size_t)(n1 +     C) * C + kb), aiz1);
    ain1 = WMMA_BF16(ax, *(const v16bf*)(Wih + (size_t)(n1 + 2 * C) * C + kb), ain1);
    ahr1 = WMMA_BF16(ah, *(const v16bf*)(Whh + (size_t)(n1        ) * C + kb), ahr1);
    ahz1 = WMMA_BF16(ah, *(const v16bf*)(Whh + (size_t)(n1 +     C) * C + kb), ahz1);
    ahn1 = WMMA_BF16(ah, *(const v16bf*)(Whh + (size_t)(n1 + 2 * C) * C + kb), ahn1);
  }
  const float bir0 = bih[n0], biz0 = bih[C + n0], bin0 = bih[2 * C + n0];
  const float bhr0 = bhh[n0], bhz0 = bhh[C + n0], bhn0 = bhh[2 * C + n0];
  const float bir1 = bih[n1], biz1 = bih[C + n1], bin1 = bih[2 * C + n1];
  const float bhr1 = bhh[n1], bhz1 = bhh[C + n1], bhn1 = bhh[2 * C + n1];
#pragma unroll
  for (int r = 0; r < 8; ++r) {
    const int mm = row0 + r + 8 * hl;
    const size_t base = (size_t)mm * C;
    {
      const float rg = 1.0f / (1.0f + __expf(-(air0[r] + bir0 + ahr0[r] + bhr0)));
      const float zg = 1.0f / (1.0f + __expf(-(aiz0[r] + biz0 + ahz0[r] + bhz0)));
      const float ng = tanhf(ain0[r] + bin0 + rg * (ahn0[r] + bhn0));
      const float hp = Hf[base + n0];
      const float hn = (1.0f - zg) * ng + zg * hp;
      Hout[base + n0] = hn;
      Hbf[base + n0]  = f2bf(hn);
    }
    {
      const float rg = 1.0f / (1.0f + __expf(-(air1[r] + bir1 + ahr1[r] + bhr1)));
      const float zg = 1.0f / (1.0f + __expf(-(aiz1[r] + biz1 + ahz1[r] + bhz1)));
      const float ng = tanhf(ain1[r] + bin1 + rg * (ahn1[r] + bhn1));
      const float hp = Hf[base + n1];
      const float hn = (1.0f - zg) * ng + zg * hp;
      Hout[base + n1] = hn;
      Hbf[base + n1]  = f2bf(hn);
    }
  }
}

// ---------------- host orchestration ----------------
extern "C" void kernel_launch(void* const* d_in, const int* in_sizes, int n_in,
                              void* d_out, int out_size, void* d_ws, size_t ws_size,
                              hipStream_t stream) {
  const float* x   = (const float*)d_in[0];   // [S,N,C]
  const int*   ei  = (const int*)d_in[1];     // [S,2,E]
  const float* ew  = (const float*)d_in[2];   // [S,E]
  const float* Wg  = (const float*)d_in[3];   // [L,C,C]
  const float* bg  = (const float*)d_in[4];   // [L,C]
  const float* lns = (const float*)d_in[5];   // [L,C]
  const float* lnb = (const float*)d_in[6];   // [L,C]
  const float* wih = (const float*)d_in[7];   // [3C,C]
  const float* whh = (const float*)d_in[8];   // [3C,C]
  const float* bih = (const float*)d_in[9];   // [3C]
  const float* bhh = (const float*)d_in[10];  // [3C]
  float* out = (float*)d_out;                 // [N,C]

  char* ws = (char*)d_ws;
  size_t off = 0;
  auto wsalloc = [&](size_t bytes) -> char* {
    char* p = ws + off;
    off = (off + bytes + 255) & ~(size_t)255;
    return p;
  };
  unsigned short* hbf      = (unsigned short*)wsalloc((size_t)N * C * 2);
  float*          hW       = (float*)wsalloc((size_t)N * C * 4);
  float*          agg      = (float*)wsalloc((size_t)N * C * 4);
  float*          deg      = (float*)wsalloc((size_t)N * 4);
  float*          dinv     = (float*)wsalloc((size_t)N * 4);
  unsigned short* featsbf  = (unsigned short*)wsalloc((size_t)S * N * C * 2);
  float*          hstate   = (float*)wsalloc((size_t)N * C * 4);
  unsigned short* hstatebf = (unsigned short*)wsalloc((size_t)N * C * 2);
  unsigned short* wtgcn    = (unsigned short*)wsalloc((size_t)L * C * C * 2);
  unsigned short* wihbf    = (unsigned short*)wsalloc((size_t)3 * C * C * 2);
  unsigned short* whhbf    = (unsigned short*)wsalloc((size_t)3 * C * C * 2);

  const long NC = (long)N * C;
  const int b256 = 256;
  auto gblk = [](long n) { return (int)((n + 255) / 256); };

  constexpr int ROWS64  = (N / 64) * 64;            // 49984 rows via 64-row kernel
  constexpr int BLKS64  = ROWS64 / 64;              // 781
  constexpr int BLKS16T = (N - ROWS64) / 16;        // 1 tail tile

  // stage weights (once per call)
  stage_wgcn_kernel<<<gblk((long)L * C * C), b256, 0, stream>>>(Wg, wtgcn);
  cast_bf_kernel<<<gblk((long)3 * C * C), b256, 0, stream>>>(wih, wihbf, (long)3 * C * C);
  cast_bf_kernel<<<gblk((long)3 * C * C), b256, 0, stream>>>(whh, whhbf, (long)3 * C * C);

  // ---- snapshots: 3x (GEMM -> scatter -> LN/ReLU) ----
  for (int s = 0; s < S; ++s) {
    const int* rows = ei + (size_t)s * 2 * E;
    const int* cols = rows + E;
    const float* ews = ew + (size_t)s * E;

    cast_bf_kernel<<<gblk(NC), b256, 0, stream>>>(x + (size_t)s * NC, hbf, NC);

    fill_f32_kernel<<<gblk(N), b256, 0, stream>>>(deg, N, 0.0f);
    degree_kernel<<<gblk(E), b256, 0, stream>>>(cols, ews, deg);
    dinv_kernel<<<gblk(N), b256, 0, stream>>>(deg, dinv);

    for (int l = 0; l < L; ++l) {
      const unsigned short* Wl = wtgcn + (size_t)l * C * C;
      gcn_gemm64_kernel<<<BLKS64, b256, 0, stream>>>(hbf, Wl, hW, 0);
      gcn_gemm16_kernel<<<BLKS16T, b256, 0, stream>>>(hbf, Wl, hW, ROWS64);
      fill_f32_kernel<<<gblk(NC), b256, 0, stream>>>(agg, NC, 0.0f);
      agg_kernel<<<E / 4, b256, 0, stream>>>(rows, cols, ews, dinv, hW, agg);
      unsigned short* dst = (l == L - 1) ? (featsbf + (size_t)s * NC) : hbf;
      ln_relu_kernel<<<N, b256, 0, stream>>>(agg, hW, dinv, bg + (size_t)l * C,
                                             lns + (size_t)l * C, lnb + (size_t)l * C, dst);
    }
  }

  // ---- GRU over S steps (h0 = 0), fused GEMMs + gates ----
  fill_f32_kernel<<<gblk(NC), b256, 0, stream>>>(hstate, NC, 0.0f);
  fill_u16_kernel<<<gblk(NC), b256, 0, stream>>>(hstatebf, NC, (unsigned short)0);
  for (int t = 0; t < S; ++t) {
    float* hout = (t == S - 1) ? out : hstate;
    gru_step_kernel<<<N / 16, b256, 0, stream>>>(featsbf + (size_t)t * NC, hstatebf, hstate,
                                                 wihbf, whhbf, bih, bhh, hout);
  }
}
